// Convolution_58076547777063
// MI455X (gfx1250) — matrix operand order repaired and checked by reading, not compiled, not run on hardware
//
#include <hip/hip_runtime.h>

typedef __attribute__((ext_vector_type(16))) _Float16 v16h;
typedef __attribute__((ext_vector_type(8)))  _Float16 v8h;
typedef __attribute__((ext_vector_type(8)))  float    v8f;
typedef __attribute__((ext_vector_type(4)))  float    v4f;

#define HC 128
#define NHEAD 4
#define NEG_SLOPE 0.2f

// ---- monotone float<->uint encoding for atomicMax on signed floats ----
__device__ __forceinline__ unsigned fenc(float f) {
    unsigned u = __float_as_uint(f);
    return (u & 0x80000000u) ? ~u : (u | 0x80000000u);
}
__device__ __forceinline__ float fdec(unsigned u) {
    return (u & 0x80000000u) ? __uint_as_float(u & 0x7FFFFFFFu) : __uint_as_float(~u);
}
// enc(-inf) = ~0xFF800000 = 0x007FFFFF
#define ENC_NEG_INF 0x007FFFFFu

// ------------------------- utility kernels -------------------------
__global__ void k_fill_f32(float* __restrict__ p, float v, long n) {
    long i = (long)blockIdx.x * blockDim.x + threadIdx.x;
    if (i < n) p[i] = v;
}
__global__ void k_fill_u32(unsigned* __restrict__ p, unsigned v, long n) {
    long i = (long)blockIdx.x * blockDim.x + threadIdx.x;
    if (i < n) p[i] = v;
}
__global__ void k_cvt_f2h(const float* __restrict__ in, _Float16* __restrict__ out, long n) {
    long i = (long)blockIdx.x * blockDim.x + threadIdx.x;
    if (i < n) out[i] = (_Float16)in[i];
}

// degree / sum of incoming edge attrs (for self-loop fill_value='mean')
__global__ void k_deg(const int* __restrict__ dst, const float* __restrict__ attr,
                      float* __restrict__ sums, float* __restrict__ cnts, int E) {
    int e = blockIdx.x * blockDim.x + threadIdx.x;
    if (e >= E) return;
    int d = dst[e];
    atomicAdd(sums + d, attr[e]);
    atomicAdd(cnts + d, 1.0f);
}
__global__ void k_loop_attr(float* __restrict__ la, const float* __restrict__ sums,
                            const float* __restrict__ cnts, int n) {
    int i = blockIdx.x * blockDim.x + threadIdx.x;
    if (i >= n) return;
    float c = cnts[i];
    la[i] = (c > 0.0f) ? (sums[i] / fmaxf(c, 1.0f)) : 0.0f;
}

// Pack W (K=128 x N=128, row major f32) into WMMA B-fragment order (f16).
// layout: pB[ ((nt*4+ks)*32 + lane)*16 + i ] ; K = ks*32 + (i<8 ? 8g+i : 16+8g+(i-8)), g=(lane>=16), col = nt*16 + lane%16
__global__ void k_packW(const float* __restrict__ W, _Float16* __restrict__ pB) {
    int idx = blockIdx.x * blockDim.x + threadIdx.x;
    if (idx >= 8 * 4 * 32 * 16) return;
    int i    = idx & 15;
    int lane = (idx >> 4) & 31;
    int ks   = (idx >> 9) & 3;
    int nt   = idx >> 11;
    int col  = nt * 16 + (lane & 15);
    int g    = lane >> 4;
    int k    = ks * 32 + (i < 8 ? 8 * g + i : 16 + 8 * g + (i - 8));
    pB[idx] = (_Float16)W[(size_t)k * HC + col];
}

// ------------------------- WMMA GEMM -------------------------
// C[M x 128] = A(f16, M x 128) @ Wpacked + bias ; one wave per 16-row strip.
// A fragment (16x32 f16, ISA 7.12.2): lane row = lane%16, g=lane/16,
//   halves 0..7  -> K = kb + 8g + i ; halves 8..15 -> K = kb + 16 + 8g + (i-8)
__global__ void k_gemm(const _Float16* __restrict__ Ah, const _Float16* __restrict__ pB,
                       const float* __restrict__ bias, float* __restrict__ Cf,
                       _Float16* __restrict__ Ch, int relu, int Mtiles) {
    int wave = (int)((blockIdx.x * blockDim.x + threadIdx.x) >> 5);
    int lane = threadIdx.x & 31;
    if (wave >= Mtiles) return;                 // wave-uniform; EXEC stays all-1
    int g = lane >> 4;
    const _Float16* arow = Ah + (size_t)(wave * 16 + (lane & 15)) * HC;

    v8f acc[8];
#pragma unroll
    for (int nt = 0; nt < 8; ++nt) acc[nt] = (v8f)0.0f;

#pragma unroll
    for (int ks = 0; ks < 4; ++ks) {
        int kb = ks * 32;
        v8h a0 = *(const v8h*)(arow + kb + 8 * g);
        v8h a1 = *(const v8h*)(arow + kb + 16 + 8 * g);
        v16h A;
#pragma unroll
        for (int i = 0; i < 8; ++i) { A[i] = a0[i]; A[8 + i] = a1[i]; }
#pragma unroll
        for (int nt = 0; nt < 8; ++nt) {
            const _Float16* bp = pB + (size_t)(((nt * 4 + ks) * 32 + lane) * 16);
            v8h b0 = *(const v8h*)bp;
            v8h b1 = *(const v8h*)(bp + 8);
            v16h B;
#pragma unroll
            for (int i = 0; i < 8; ++i) { B[i] = b0[i]; B[8 + i] = b1[i]; }
            acc[nt] = __builtin_amdgcn_wmma_f32_16x16x32_f16(
                false, A, false, B, (short)0, acc[nt], false, false);
        }
    }
    // C/D layout: VGPR r holds row r + 8g, col = lane%16 within tile
#pragma unroll
    for (int nt = 0; nt < 8; ++nt) {
        int col = nt * 16 + (lane & 15);
        float bv = bias[col];
#pragma unroll
        for (int r = 0; r < 8; ++r) {
            int m = wave * 16 + r + 8 * g;
            float v = acc[nt][r] + bv;
            if (relu) v = v > 0.0f ? v : 0.0f;
            if (Cf) Cf[(size_t)m * HC + col] = v;
            if (Ch) Ch[(size_t)m * HC + col] = (_Float16)v;
        }
    }
}

// ------------------------- edge phase -------------------------
// wave per edge (virtual edges e>=E are self loops). lane owns channels lane*4..lane*4+3,
// head = lane/8 (32 channels per head -> 8 lanes).
__global__ void k_edge_logits(const int* __restrict__ src, const int* __restrict__ dst,
                              const float* __restrict__ attr, const float* __restrict__ lattr,
                              const float* __restrict__ xl, const float* __restrict__ xr,
                              const float* __restrict__ We, const float* __restrict__ att,
                              float* __restrict__ logits, unsigned* __restrict__ mx,
                              int E, int Nn) {
    int e = (int)((blockIdx.x * blockDim.x + threadIdx.x) >> 5);
    int lane = threadIdx.x & 31;
    if (e >= E + Nn) return;
    int s, d; float a;
    if (e < E) { s = src[e]; d = dst[e]; a = attr[e]; }
    else       { s = d = e - E; a = lattr[s]; }
    int c0 = lane * 4;
    v4f vl = *(const v4f*)(xl + (size_t)s * HC + c0);
    v4f vr = *(const v4f*)(xr + (size_t)d * HC + c0);
    v4f vw = *(const v4f*)(We + c0);
    v4f va = *(const v4f*)(att + c0);
    float p = 0.0f;
#pragma unroll
    for (int j = 0; j < 4; ++j) {
        float m = vl[j] + vr[j] + a * vw[j];
        m = m > 0.0f ? m : NEG_SLOPE * m;
        p += m * va[j];
    }
    p += __shfl_xor(p, 1);
    p += __shfl_xor(p, 2);
    p += __shfl_xor(p, 4);
    if ((lane & 7) == 0) {
        int h = lane >> 3;
        logits[(size_t)e * NHEAD + h] = p;
        atomicMax(mx + (size_t)d * NHEAD + h, fenc(p));
    }
}

__global__ void k_edge_exp(const int* __restrict__ dst, float* __restrict__ logits,
                           const unsigned* __restrict__ mx, float* __restrict__ sm,
                           int E, int Nn) {
    int e = blockIdx.x * blockDim.x + threadIdx.x;
    if (e >= E + Nn) return;
    int d = (e < E) ? dst[e] : e - E;
#pragma unroll
    for (int h = 0; h < NHEAD; ++h) {
        float l = logits[(size_t)e * NHEAD + h];
        float m = fdec(mx[(size_t)d * NHEAD + h]);
        float ev = __expf(l - m);
        logits[(size_t)e * NHEAD + h] = ev;
        atomicAdd(sm + (size_t)d * NHEAD + h, ev);
    }
}

__global__ void k_edge_aggr(const int* __restrict__ src, const int* __restrict__ dst,
                            const float* __restrict__ logits, const float* __restrict__ sm,
                            const float* __restrict__ xl, float* __restrict__ agg,
                            int E, int Nn) {
    int e = (int)((blockIdx.x * blockDim.x + threadIdx.x) >> 5);
    int lane = threadIdx.x & 31;
    if (e >= E + Nn) return;
    int s, d;
    if (e < E) { s = src[e]; d = dst[e]; }
    else       { s = d = e - E; }
    int h = lane >> 3;
    float alpha = logits[(size_t)e * NHEAD + h] /
                  (sm[(size_t)d * NHEAD + h] + 1e-16f);
    int c0 = lane * 4;
    v4f vl = *(const v4f*)(xl + (size_t)s * HC + c0);
#pragma unroll
    for (int j = 0; j < 4; ++j)
        atomicAdd(agg + (size_t)d * HC + c0 + j, vl[j] * alpha);
}

// out-bias + ReLU, emit f16 for the next GEMM input
__global__ void k_finish(const float* __restrict__ agg, const float* __restrict__ bo,
                         _Float16* __restrict__ hh, long n) {
    long i = (long)blockIdx.x * blockDim.x + threadIdx.x;
    if (i >= n) return;
    float v = agg[i] + bo[(int)(i & (HC - 1))];
    hh[i] = (_Float16)(v > 0.0f ? v : 0.0f);
}

// ------------------------- host orchestration -------------------------
extern "C" void kernel_launch(void* const* d_in, const int* in_sizes, int n_in,
                              void* d_out, int out_size, void* d_ws, size_t ws_size,
                              hipStream_t stream) {
    const int N  = in_sizes[0] / HC;     // 50000
    const int E  = in_sizes[2];          // 800000
    const int Et = E + N;
    const int Mt = N / 16;               // 3125 (exact)

    const float* x     = (const float*)d_in[0];
    const int*   src   = (const int*)d_in[1];
    const int*   dst   = src + E;
    const float* eat   = (const float*)d_in[2];
    const float* Wl1 = (const float*)d_in[3],  *bl1 = (const float*)d_in[4];
    const float* Wr1 = (const float*)d_in[5],  *br1 = (const float*)d_in[6];
    const float* We1 = (const float*)d_in[7],  *at1 = (const float*)d_in[8];
    const float* bo1 = (const float*)d_in[9];
    const float* Wl2 = (const float*)d_in[10], *bl2 = (const float*)d_in[11];
    const float* Wr2 = (const float*)d_in[12], *br2 = (const float*)d_in[13];
    const float* We2 = (const float*)d_in[14], *at2 = (const float*)d_in[15];
    const float* bo2 = (const float*)d_in[16];
    const float* Wlin = (const float*)d_in[17], *blin = (const float*)d_in[18];

    char* ws = (char*)d_ws;
    size_t off = 0;
    auto take = [&](size_t bytes) -> char* {
        char* p = ws + off;
        off = (off + bytes + 255) & ~(size_t)255;
        return p;
    };
    _Float16* xh  = (_Float16*)take((size_t)N * HC * 2);
    _Float16* hh  = (_Float16*)take((size_t)N * HC * 2);
    _Float16* pBl = (_Float16*)take(16384 * 2);
    _Float16* pBr = (_Float16*)take(16384 * 2);
    float* xl     = (float*)take((size_t)N * HC * 4);
    float* xr     = (float*)take((size_t)N * HC * 4);
    float* agg    = (float*)take((size_t)N * HC * 4);
    float* logits = (float*)take((size_t)Et * NHEAD * 4);
    unsigned* mx  = (unsigned*)take((size_t)N * NHEAD * 4);
    float* sm     = (float*)take((size_t)N * NHEAD * 4);
    float* sums   = (float*)take((size_t)N * 4);
    float* cnts   = (float*)take((size_t)N * 4);
    float* lattr  = (float*)take((size_t)N * 4);
    (void)ws_size; (void)n_in; (void)out_size;

    auto nb = [](long n, int t) { return (unsigned)((n + t - 1) / t); };
    const int T = 256;
    const unsigned gemmBlocks = nb((long)Mt * 32, T);
    const unsigned edgeBlocks = nb((long)Et * 32, T);

    // self-loop mean attrs
    k_fill_f32<<<nb(N, T), T, 0, stream>>>(sums, 0.0f, N);
    k_fill_f32<<<nb(N, T), T, 0, stream>>>(cnts, 0.0f, N);
    k_deg<<<nb(E, T), T, 0, stream>>>(dst, eat, sums, cnts, E);
    k_loop_attr<<<nb(N, T), T, 0, stream>>>(lattr, sums, cnts, N);
    k_cvt_f2h<<<nb((long)N * HC, T), T, 0, stream>>>(x, xh, (long)N * HC);

    struct Layer {
        const float *Wl, *bl, *Wr, *br, *We, *att, *bo;
        const _Float16* in; _Float16* out;
    } L[2] = {
        { Wl1, bl1, Wr1, br1, We1, at1, bo1, xh, hh },
        { Wl2, bl2, Wr2, br2, We2, at2, bo2, hh, xh },   // reuse xh for h2
    };

    for (int l = 0; l < 2; ++l) {
        k_packW<<<nb(16384, T), T, 0, stream>>>(L[l].Wl, pBl);
        k_packW<<<nb(16384, T), T, 0, stream>>>(L[l].Wr, pBr);
        k_gemm<<<gemmBlocks, T, 0, stream>>>(L[l].in, pBl, L[l].bl, xl, (_Float16*)nullptr, 0, Mt);
        k_gemm<<<gemmBlocks, T, 0, stream>>>(L[l].in, pBr, L[l].br, xr, (_Float16*)nullptr, 0, Mt);

        k_fill_u32<<<nb((long)N * NHEAD, T), T, 0, stream>>>(mx, ENC_NEG_INF, (long)N * NHEAD);
        k_fill_f32<<<nb((long)N * NHEAD, T), T, 0, stream>>>(sm, 0.0f, (long)N * NHEAD);
        k_fill_f32<<<nb((long)N * HC, T), T, 0, stream>>>(agg, 0.0f, (long)N * HC);

        k_edge_logits<<<edgeBlocks, T, 0, stream>>>(src, dst, eat, lattr, xl, xr,
                                                    L[l].We, L[l].att, logits, mx, E, N);
        k_edge_exp<<<nb(Et, T), T, 0, stream>>>(dst, logits, mx, sm, E, N);
        k_edge_aggr<<<edgeBlocks, T, 0, stream>>>(src, dst, logits, sm, xl, agg, E, N);
        k_finish<<<nb((long)N * HC, T), T, 0, stream>>>(agg, L[l].bo, L[l].out, (long)N * HC);
    }

    // final linear: relu(h2 @ Wlin + blin) -> d_out (f32)
    k_packW<<<nb(16384, T), T, 0, stream>>>(Wlin, pBl);
    k_gemm<<<gemmBlocks, T, 0, stream>>>(xh, pBl, blin, (float*)d_out, (_Float16*)nullptr, 1, Mt);
}
